// MultiWaveletTransform_89833535963429
// MI455X (gfx1250) — matrix operand order repaired and checked by reading, not compiled, not run on hardware
//
#include <hip/hip_runtime.h>
#include <math.h>

// ---------------------------------------------------------------------------
// MultiWaveletTransform for MI455X (gfx1250).
// The reference zero-pads all rFFT bins >= 16 before irfft, so the "FFT" is a
// 16-bin truncated DFT == skinny GEMM.  All large contractions (projections,
// forward DFT, per-frequency complex spectral mix, inverse DFT) run on
// v_wmma_f32_16x16x32_f16 (f16 operands, f32 accumulate, wave32 layouts).
// LDS staging: unconditional float4 global loads -> packed f16 -> b128 LDS
// stores; B tiles stored transposed so each lane's fragment is one contiguous
// 32B LDS read.
// ---------------------------------------------------------------------------

typedef __attribute__((ext_vector_type(16))) _Float16 v16h;
typedef __attribute__((ext_vector_type(8)))  _Float16 h8;
typedef __attribute__((ext_vector_type(8)))  float    v8f;

#define WMMA_F16(a, b, c) \
  __builtin_amdgcn_wmma_f32_16x16x32_f16(false, (a), false, (b), (short)0, (c), false, false)

static __device__ __forceinline__ h8 cvt8(float4 a, float4 b) {
  h8 r;
  r[0] = (_Float16)a.x; r[1] = (_Float16)a.y; r[2] = (_Float16)a.z; r[3] = (_Float16)a.w;
  r[4] = (_Float16)b.x; r[5] = (_Float16)b.y; r[6] = (_Float16)b.z; r[7] = (_Float16)b.w;
  return r;
}

// A fragment (16x32 f16, MxK, row-major LDS, ld=32):
// element j -> k = j + 16*hi: two contiguous 16B runs.
static __device__ __forceinline__ v16h a_frag(const _Float16* __restrict__ row, int hi) {
  h8 lo = *(const h8*)(row + 16 * hi);
  h8 hh = *(const h8*)(row + 16 * hi + 8);
  return __builtin_shufflevector(lo, hh, 0, 1, 2, 3, 4, 5, 6, 7, 8, 9, 10, 11, 12, 13, 14, 15);
}

// B fragment (32x16 f16, KxN) from TRANSPOSED LDS tile BsT[n][k] (ld=32):
// element j -> k = j + 16*hi: one contiguous 32B run per lane.
static __device__ __forceinline__ v16h b_frag(const _Float16* __restrict__ BsT, int col, int hi) {
  return *(const v16h*)(BsT + col * 32 + 16 * hi);
}

// ---------------------------------------------------------------------------
// GEMM:  C[M,N] = A[M,K] @ W[N,K]^T + bias[N]   (exact tiles: M%128==0, N%64==0, K%32==0)
// block tile 128(M) x 64(N), 8 waves (4x2 of 32x32), K step 32.
// ---------------------------------------------------------------------------
__global__ __launch_bounds__(256) void gemm_rowmajor_wtrans(
    const float* __restrict__ A, const float* __restrict__ W,
    const float* __restrict__ bias, float* __restrict__ C, int M, int N, int Kd) {
  __shared__ __align__(32) _Float16 As[128][32];
  __shared__ __align__(32) _Float16 BsT[64][32];
  int tid = threadIdx.x;
  int lane = tid & 31, wave = tid >> 5;
  int hi = lane >> 4, ln = lane & 15;
  int wm = (wave >> 1) * 32, wn = (wave & 1) * 32;
  int bm = blockIdx.y * 128, bn = blockIdx.x * 64;
  v8f acc[2][2] = {};
  for (int k0 = 0; k0 < Kd; k0 += 32) {
    if (k0 + 32 < Kd)  // pull next A tile toward the caches (global_prefetch_b8)
      __builtin_prefetch(&A[(size_t)(bm + (tid >> 1)) * Kd + k0 + 32], 0, 1);
    {  // A tile: thread -> (row = tid>>1, 16-wide k chunk)
      int r = tid >> 1, kb = (tid & 1) * 16;
      const float4* s4 = (const float4*)(A + (size_t)(bm + r) * Kd + k0 + kb);
      float4 f0 = s4[0], f1 = s4[1], f2 = s4[2], f3 = s4[3];
      *(h8*)&As[r][kb] = cvt8(f0, f1);
      *(h8*)&As[r][kb + 8] = cvt8(f2, f3);
    }
    {  // B tile (transposed in LDS): thread -> (n = tid>>2, 8-wide k chunk)
      int n = tid >> 2, kb = (tid & 3) * 8;
      const float4* s4 = (const float4*)(W + (size_t)(bn + n) * Kd + k0 + kb);
      float4 f0 = s4[0], f1 = s4[1];
      *(h8*)&BsT[n][kb] = cvt8(f0, f1);
    }
    __syncthreads();
#pragma unroll
    for (int mi = 0; mi < 2; ++mi) {
      v16h a = a_frag(&As[wm + mi * 16 + ln][0], hi);
#pragma unroll
      for (int ni = 0; ni < 2; ++ni) {
        v16h b = b_frag(&BsT[0][0], wn + ni * 16 + ln, hi);
        acc[mi][ni] = WMMA_F16(a, b, acc[mi][ni]);
      }
    }
    __syncthreads();
  }
#pragma unroll
  for (int mi = 0; mi < 2; ++mi)
#pragma unroll
    for (int ni = 0; ni < 2; ++ni)
#pragma unroll
      for (int r = 0; r < 8; ++r) {
        int m = bm + wm + mi * 16 + r + 8 * hi;
        int n = bn + wn + ni * 16 + ln;
        C[(size_t)m * N + n] = acc[mi][ni][r] + bias[n];
      }
}

// circular pad: v[b, 3072+j, :] = v[b, j, :]  for j < 1024 (rows stride 4096)
__global__ void pad_kernel(float* __restrict__ v) {
  int g = blockIdx.x * blockDim.x + threadIdx.x;
  if (g >= 4 * 1024 * 512) return;
  int ch = g & 511;
  int j = (g >> 9) & 1023;
  int b = g >> 19;
  v[(size_t)b * 4096 * 512 + (size_t)(3072 + j) * 512 + ch] =
      v[(size_t)b * 4096 * 512 + (size_t)j * 512 + ch];
}

// ---------------------------------------------------------------------------
// Wavelet analysis:  xa = [v[2t], v[2t+1]] (per C, 2K=16);  d = xa@ec_d, s = xa@ec_s
// ---------------------------------------------------------------------------
__global__ __launch_bounds__(256) void analysis_kernel(
    const float* __restrict__ vin, float* __restrict__ dOut, float* __restrict__ sOut,
    const float* __restrict__ ecs, const float* __restrict__ ecd, int Lh, int inStrideB,
    int outStrideB) {
  __shared__ float fs[128], fd[128];
  if (threadIdx.x < 128) { fs[threadIdx.x] = ecs[threadIdx.x]; fd[threadIdx.x] = ecd[threadIdx.x]; }
  __syncthreads();
  long g = (long)blockIdx.x * blockDim.x + threadIdx.x;
  long total = (long)4 * Lh * 64;
  if (g >= total) return;
  int c = (int)(g & 63);
  int t = (int)((g >> 6) % Lh);
  int b = (int)(g / ((long)Lh * 64));
  const float* e0 = vin + (size_t)b * inStrideB + (size_t)(2 * t) * 512 + c * 8;
  const float* e1 = e0 + 512;
  float xa[16];
#pragma unroll
  for (int j = 0; j < 8; ++j) { xa[j] = e0[j]; xa[8 + j] = e1[j]; }
  float dv[8] = {}, sv[8] = {};
#pragma unroll
  for (int j = 0; j < 16; ++j)
#pragma unroll
    for (int ko = 0; ko < 8; ++ko) {
      dv[ko] += xa[j] * fd[j * 8 + ko];
      sv[ko] += xa[j] * fs[j * 8 + ko];
    }
  float* dp = dOut + (size_t)b * outStrideB + (size_t)t * 512 + c * 8;
  float* sp = sOut + (size_t)b * outStrideB + (size_t)t * 512 + c * 8;
#pragma unroll
  for (int ko = 0; ko < 8; ++ko) { dp[ko] = dv[ko]; sp[ko] = sv[ko]; }
}

// ---------------------------------------------------------------------------
// Forward truncated DFT (16 bins) via WMMA:
// Out[32,512] = Basis[32,Lh] @ In_b[Lh,512]; row 2f = cos, row 2f+1 = -sin.
// grid = (8 n-tiles, 4 batch), 128 threads (4 waves: 2 M-tiles x 2 N-tiles).
// ---------------------------------------------------------------------------
__global__ __launch_bounds__(128) void dft_fwd_kernel(const float* __restrict__ in,
                                                      float* __restrict__ hat, int Lh,
                                                      int inStrideB, int lvl) {
  __shared__ __align__(32) _Float16 As[32][32];
  __shared__ __align__(32) _Float16 BsT[64][32];
  int tid = threadIdx.x, lane = tid & 31, wave = tid >> 5;
  int hi = lane >> 4, ln = lane & 15;
  int wm = (wave & 1) * 16, wn = (wave >> 1) * 32;
  int b = blockIdx.y, bn = blockIdx.x * 64;
  float invL = 1.0f / (float)Lh;
  v8f acc[2] = {};
  for (int t0 = 0; t0 < Lh; t0 += 32) {
#pragma unroll
    for (int e = tid; e < 1024; e += 128) {  // basis tile (pure VALU)
      int r = e >> 5, tt = e & 31;
      int t = t0 + tt;
      int f = r >> 1;
      float val = 0.f;
      if (t < Lh) {
        float ang = 6.28318530717958647692f * (float)((f * t) % Lh) * invL;
        val = (r & 1) ? -__sinf(ang) : __cosf(ang);
      }
      As[r][tt] = (_Float16)val;
    }
    {  // data tile, transposed: thread -> (c = tid&63, 16-wide t chunk)
      int c = tid & 63, half = tid >> 6;
      _Float16 vals[16];
#pragma unroll
      for (int j = 0; j < 16; ++j) {
        int t = t0 + 16 * half + j;
        int tc = (t < Lh) ? t : (Lh - 1);  // clamp: load unconditionally, select value
        float v = in[(size_t)b * inStrideB + (size_t)tc * 512 + bn + c];
        vals[j] = (_Float16)((t < Lh) ? v : 0.f);
      }
#pragma unroll
      for (int j = 0; j < 16; ++j) BsT[c][16 * half + j] = vals[j];
    }
    __syncthreads();
    v16h a = a_frag(&As[wm + ln][0], hi);
#pragma unroll
    for (int ni = 0; ni < 2; ++ni) {
      v16h bb = b_frag(&BsT[0][0], wn + ni * 16 + ln, hi);
      acc[ni] = WMMA_F16(a, bb, acc[ni]);
    }
    __syncthreads();
  }
#pragma unroll
  for (int ni = 0; ni < 2; ++ni)
#pragma unroll
    for (int r = 0; r < 8; ++r) {
      int m = wm + r + 8 * hi;  // 0..31
      int f = m >> 1, reim = m & 1;
      int n = bn + wn + ni * 16 + ln;
      hat[((((size_t)lvl * 16 + f) * 4 + b) * 2 + reim) * 512 + n] = acc[ni][r];
    }
}

// ---------------------------------------------------------------------------
// Per-frequency spectral mix, all 11 levels x 4 batches packed into M=48 (44 used).
// Complex matmul as real GEMM: X = [Dre Dim Vre Vim] (K=2048), W stacked with signs.
// variant 0: Udhat = Dhat*A + Vhat*B ; variant 1: Ushat = Dhat*C (K=1024)
// grid = (16 n-tiles over N=1024 (re|im), 16 freq, 2 variant), 256 thr; waves 0..5 compute.
// ---------------------------------------------------------------------------
__global__ __launch_bounds__(256) void spectral_kernel(
    const float* __restrict__ Dhat, const float* __restrict__ Vhat,
    const float* __restrict__ Ar, const float* __restrict__ Ai, const float* __restrict__ Br,
    const float* __restrict__ Bi, const float* __restrict__ Cr, const float* __restrict__ Ci,
    float* __restrict__ Udhat, float* __restrict__ Ushat) {
  __shared__ __align__(32) _Float16 As[48][32];
  __shared__ __align__(32) _Float16 BsT[64][32];
  int tid = threadIdx.x, lane = tid & 31, wave = tid >> 5;
  int hi = lane >> 4, ln = lane & 15;
  int wm = (wave % 3) * 16, wn = (wave / 3) * 32;  // used by waves 0..5
  int f = blockIdx.y, bn = blockIdx.x * 64;
  int variant = blockIdx.z;
  int Kd = variant ? 1024 : 2048;
  v8f acc[2] = {};
  for (int k0 = 0; k0 < Kd; k0 += 32) {
    if (tid < 192) {  // X tile: thread -> (m = tid>>2, 8-wide k chunk); rows 44..47 zero
      int m = tid >> 2, kb = (tid & 3) * 8;
      int l = m >> 2, bq = m & 3;
      int k = k0 + kb;
      int q = k >> 9, i = k & 511;  // q: 0=Dre 1=Dim 2=Vre 3=Vim
      int lc = (l < 11) ? l : 10;
      const float* src = (q < 2) ? Dhat : Vhat;
      const float4* s4 =
          (const float4*)&src[((((size_t)lc * 16 + f) * 4 + bq) * 2 + (q & 1)) * 512 + i];
      float4 f0 = s4[0], f1 = s4[1];
      h8 p = cvt8(f0, f1);
      if (l >= 11) { h8 z = {}; p = z; }
      *(h8*)&As[m][kb] = p;
    }
#pragma unroll
    for (int e = tid; e < 2048; e += 256) {  // W tile gather (weights are (i,o,f) strided)
      int kk = e >> 6, c = e & 63;
      int k = k0 + kk;
      int q = k >> 9, i = k & 511;
      int n = bn + c;
      int o = n & 511, isIm = n >> 9;
      const float* Wr;
      const float* Wi;
      if (variant) { Wr = Cr; Wi = Ci; }
      else if (q < 2) { Wr = Ar; Wi = Ai; }
      else { Wr = Br; Wi = Bi; }
      size_t idx = ((size_t)i * 512 + o) * 16 + f;
      float val;
      if (isIm == 0) val = (q & 1) ? -Wi[idx] : Wr[idx];
      else           val = (q & 1) ?  Wr[idx] : Wi[idx];
      BsT[c][kk] = (_Float16)val;
    }
    __syncthreads();
    if (wave < 6) {
      v16h a = a_frag(&As[wm + ln][0], hi);
#pragma unroll
      for (int ni = 0; ni < 2; ++ni) {
        v16h bb = b_frag(&BsT[0][0], wn + ni * 16 + ln, hi);
        acc[ni] = WMMA_F16(a, bb, acc[ni]);
      }
    }
    __syncthreads();
  }
  if (wave >= 6) return;
  float* out = variant ? Ushat : Udhat;
#pragma unroll
  for (int ni = 0; ni < 2; ++ni)
#pragma unroll
    for (int r = 0; r < 8; ++r) {
      int m = wm + r + 8 * hi;
      int l = m >> 2, b = m & 3;
      if (l >= 11) continue;
      int n = bn + wn + ni * 16 + ln;
      int o = n & 511, isIm = n >> 9;
      out[((((size_t)l * 16 + f) * 4 + b) * 2 + isIm) * 512 + o] = acc[ni][r];
    }
}

// ---------------------------------------------------------------------------
// Inverse truncated DFT via WMMA: Out_b[Lh,512] = iBasis[Lh,32] @ Hat_b[32,512].
// iBasis folds 1/L, the 2x for non-DC bins and the Nyquist weight-1 case.
// grid = (8 n-tiles, 4 batch, ceil(Lh/32)), 128 threads, single K step.
// ---------------------------------------------------------------------------
__global__ __launch_bounds__(128) void idft_kernel(const float* __restrict__ hat,
                                                   float* __restrict__ out, int Lh,
                                                   int outStrideB, int lvl, int lEff) {
  __shared__ __align__(32) _Float16 As[32][32];
  __shared__ __align__(32) _Float16 BsT[64][32];
  int tid = threadIdx.x, lane = tid & 31, wave = tid >> 5;
  int hi = lane >> 4, ln = lane & 15;
  int wm = (wave & 1) * 16, wn = (wave >> 1) * 32;
  int b = blockIdx.y, bn = blockIdx.x * 64;
  int m0 = blockIdx.z * 32;
  float invL = 1.0f / (float)Lh;
#pragma unroll
  for (int e = tid; e < 1024; e += 128) {  // inverse basis (pure VALU)
    int tt = e >> 5, r = e & 31;
    int t = m0 + tt;
    int fq = r >> 1;
    float val = 0.f;
    if (t < Lh && fq < lEff) {
      float ang = 6.28318530717958647692f * (float)((fq * t) % Lh) * invL;
      if (r & 1) val = -2.f * invL * __sinf(ang);
      else {
        float w = (fq == 0 || 2 * fq == Lh) ? invL : 2.f * invL;
        val = w * __cosf(ang);
      }
    }
    As[tt][r] = (_Float16)val;
  }
  {  // hat tile, transposed: thread -> (c = tid&63, 16-wide row chunk)
    int c = tid & 63, half = tid >> 6;
    _Float16 vals[16];
#pragma unroll
    for (int j = 0; j < 16; ++j) {
      int r = 16 * half + j;
      vals[j] = (_Float16)
          hat[((((size_t)lvl * 16 + (r >> 1)) * 4 + b) * 2 + (r & 1)) * 512 + bn + c];
    }
#pragma unroll
    for (int j = 0; j < 16; ++j) BsT[c][16 * half + j] = vals[j];
  }
  __syncthreads();
  v16h a = a_frag(&As[wm + ln][0], hi);
  v8f acc[2] = {};
#pragma unroll
  for (int ni = 0; ni < 2; ++ni) {
    v16h bb = b_frag(&BsT[0][0], wn + ni * 16 + ln, hi);
    acc[ni] = WMMA_F16(a, bb, acc[ni]);
  }
#pragma unroll
  for (int ni = 0; ni < 2; ++ni)
#pragma unroll
    for (int r = 0; r < 8; ++r) {
      int t = m0 + wm + r + 8 * hi;
      if (t >= Lh) continue;
      int n = bn + wn + ni * 16 + ln;
      out[(size_t)b * outStrideB + (size_t)t * 512 + n] = acc[ni][r];
    }
}

// coarsest-scale transform: out = v @ T0_w^T + T0_b   (v: B x 2 x 64 x 8, tight)
__global__ void t0_kernel(const float* __restrict__ vin, const float* __restrict__ T0w,
                          const float* __restrict__ T0b, float* __restrict__ vout) {
  int g = blockIdx.x * blockDim.x + threadIdx.x;
  if (g >= 4 * 2 * 64 * 8) return;
  int ko = g & 7;
  int rest = g >> 3;
  int c = rest & 63;
  int bt = rest >> 6;  // (b*2 + t), rows tight
  const float* vp = vin + (size_t)bt * 512 + c * 8;
  float s = T0b[ko];
#pragma unroll
  for (int j = 0; j < 8; ++j) s += vp[j] * T0w[ko * 8 + j];
  vout[(size_t)bt * 512 + c * 8 + ko] = s;
}

// reconstruction: s = v + Us; ve = [s, Ud]; xe = ve@rc_e, xo = ve@rc_o; interleave
__global__ __launch_bounds__(256) void recon_kernel(
    const float* __restrict__ vin, const float* __restrict__ Ud, const float* __restrict__ Us,
    float* __restrict__ vout, const float* __restrict__ rce, const float* __restrict__ rco,
    int Lh, int inStrideB, int outStrideB, int udStrideB) {
  __shared__ float fe[128], fo[128];
  if (threadIdx.x < 128) { fe[threadIdx.x] = rce[threadIdx.x]; fo[threadIdx.x] = rco[threadIdx.x]; }
  __syncthreads();
  long g = (long)blockIdx.x * blockDim.x + threadIdx.x;
  long total = (long)4 * Lh * 64;
  if (g >= total) return;
  int c = (int)(g & 63);
  int t = (int)((g >> 6) % Lh);
  int b = (int)(g / ((long)Lh * 64));
  const float* vp = vin + (size_t)b * inStrideB + (size_t)t * 512 + c * 8;
  const float* up = Us + (size_t)b * udStrideB + (size_t)t * 512 + c * 8;
  const float* dp = Ud + (size_t)b * udStrideB + (size_t)t * 512 + c * 8;
  float ve[16];
#pragma unroll
  for (int j = 0; j < 8; ++j) { ve[j] = vp[j] + up[j]; ve[8 + j] = dp[j]; }
  float xe[8] = {}, xo[8] = {};
#pragma unroll
  for (int j = 0; j < 16; ++j)
#pragma unroll
    for (int ko = 0; ko < 8; ++ko) {
      xe[ko] += ve[j] * fe[j * 8 + ko];
      xo[ko] += ve[j] * fo[j * 8 + ko];
    }
  float* oe = vout + (size_t)b * outStrideB + (size_t)(2 * t) * 512 + c * 8;
#pragma unroll
  for (int ko = 0; ko < 8; ++ko) { oe[ko] = xe[ko]; oe[512 + ko] = xo[ko]; }
}

// ---------------------------------------------------------------------------
extern "C" void kernel_launch(void* const* d_in, const int* in_sizes, int n_in, void* d_out,
                              int out_size, void* d_ws, size_t ws_size, hipStream_t stream) {
  (void)in_sizes; (void)n_in; (void)out_size;
  const float* x    = (const float*)d_in[0];
  const float* Lk0w = (const float*)d_in[1];
  const float* Lk0b = (const float*)d_in[2];
  const float* Lk1w = (const float*)d_in[3];
  const float* Lk1b = (const float*)d_in[4];
  const float* T0w  = (const float*)d_in[5];
  const float* T0b  = (const float*)d_in[6];
  const float* Ar   = (const float*)d_in[7];
  const float* Ai   = (const float*)d_in[8];
  const float* Br   = (const float*)d_in[9];
  const float* Bi   = (const float*)d_in[10];
  const float* Cr   = (const float*)d_in[11];
  const float* Ci   = (const float*)d_in[12];
  const float* ecs  = (const float*)d_in[13];
  const float* ecd  = (const float*)d_in[14];
  const float* rce  = (const float*)d_in[15];
  const float* rco  = (const float*)d_in[16];
  float* outp = (float*)d_out;

  float* ws = (float*)d_ws;
  size_t off = 0;
  float* vPing = ws + off; off += (size_t)4 * 4096 * 512;   // full-length v / recon ping
  float* vPong = ws + off; off += (size_t)4 * 2048 * 512;   // analysis s ping-pong
  float* dTmp  = ws + off; off += (size_t)4 * 2048 * 512;   // d scratch / recon pong
  const size_t hatSz = (size_t)11 * 16 * 4 * 2 * 512;
  float* Dhat = ws + off; off += hatSz;
  float* Vhat = ws + off; off += hatSz;
  float* Udh  = ws + off; off += hatSz;
  float* Ush  = ws + off; off += hatSz;
  float* UdB  = ws + off; off += (size_t)4 * 4094 * 512;
  float* UsB  = ws + off; off += (size_t)4 * 4094 * 512;
  if (ws_size < off * sizeof(float)) return;  // scratch too small; bail deterministically

  // 1) input projection (per batch so v gets the 4096-row padded layout)
  for (int b = 0; b < 4; ++b) {
    dim3 grid(512 / 64, 3072 / 128);
    gemm_rowmajor_wtrans<<<grid, 256, 0, stream>>>(
        x + (size_t)b * 3072 * 512, Lk0w, Lk0b, vPing + (size_t)b * 4096 * 512, 3072, 512, 512);
  }
  pad_kernel<<<(4 * 1024 * 512) / 256, 256, 0, stream>>>(vPing);

  int Lhs[11], offs[11];
  { int o = 0; for (int l = 0; l < 11; ++l) { Lhs[l] = 2048 >> l; offs[l] = o; o += Lhs[l]; } }

  // 2) analysis cascade + forward truncated DFTs
  const float* vin = vPing;
  int inStride = 4096 * 512;
  float* sBufs[2] = {vPong, vPing};
  for (int l = 0; l < 11; ++l) {
    int Lh = Lhs[l];
    float* sOut = sBufs[l & 1];
    int outStride = Lh * 512;
    long tot = (long)4 * Lh * 64;
    analysis_kernel<<<(unsigned)((tot + 255) / 256), 256, 0, stream>>>(vin, dTmp, sOut, ecs, ecd,
                                                                       Lh, inStride, outStride);
    dim3 dg(8, 4);
    dft_fwd_kernel<<<dg, 128, 0, stream>>>(dTmp, Dhat, Lh, outStride, l);
    dft_fwd_kernel<<<dg, 128, 0, stream>>>(sOut, Vhat, Lh, outStride, l);
    vin = sOut;
    inStride = outStride;
  }

  // 3) per-frequency spectral mixing (all levels batched into M)
  {
    dim3 sg(16, 16, 2);
    spectral_kernel<<<sg, 256, 0, stream>>>(Dhat, Vhat, Ar, Ai, Br, Bi, Cr, Ci, Udh, Ush);
  }

  // 4) inverse truncated DFTs -> time-domain Ud/Us per level
  for (int l = 0; l < 11; ++l) {
    int Lh = Lhs[l];
    int nf = Lh / 2 + 1;
    int lEff = nf < 16 ? nf : 16;
    dim3 ig(8, 4, (Lh + 31) / 32);
    idft_kernel<<<ig, 128, 0, stream>>>(Udh, UdB + (size_t)offs[l] * 512, Lh, 4094 * 512, l, lEff);
    idft_kernel<<<ig, 128, 0, stream>>>(Ush, UsB + (size_t)offs[l] * 512, Lh, 4094 * 512, l, lEff);
  }

  // 5) coarsest transform (level-10 s lives in vPong), output into dTmp
  t0_kernel<<<(4096 + 255) / 256, 256, 0, stream>>>(vPong, T0w, T0b, dTmp);

  // 6) reconstruction, ping-pong dTmp <-> vPing (11 steps -> ends in vPing)
  float* cur = dTmp;
  float* nxt = vPing;
  for (int i = 10; i >= 0; --i) {
    int Lh = Lhs[i];
    long tot = (long)4 * Lh * 64;
    recon_kernel<<<(unsigned)((tot + 255) / 256), 256, 0, stream>>>(
        cur, UdB + (size_t)offs[i] * 512, UsB + (size_t)offs[i] * 512, nxt, rce, rco, Lh, Lh * 512,
        2 * Lh * 512, 4094 * 512);
    float* tmp = cur; cur = nxt; nxt = tmp;
  }

  // 7) output projection from the first 3072 rows of each batch
  for (int b = 0; b < 4; ++b) {
    dim3 grid(512 / 64, 3072 / 128);
    gemm_rowmajor_wtrans<<<grid, 256, 0, stream>>>(cur + (size_t)b * 4096 * 512, Lk1w, Lk1b,
                                                   outp + (size_t)b * 3072 * 512, 3072, 512, 512);
  }
}
